// Attention_45749991637349
// MI455X (gfx1250) — matrix operand order repaired and checked by reading, not compiled, not run on hardware
//
#include <hip/hip_runtime.h>
#include <stdint.h>

#define DIM   1024
#define SEQ   1024
#define BATCH 8
#define HEADS 16
#define HD    64
#define ROWS  (BATCH * SEQ)   // 8192
#define QKVN  (3 * DIM)       // 3072

typedef __attribute__((ext_vector_type(16))) __bf16 bf16x16;
typedef __attribute__((ext_vector_type(8)))  float  floatx8;

// ---------------------------------------------------------------------------
// WMMA helpers (layouts per CDNA5 ISA §7.12.2, wave32)
// A 16x32 bf16: lane holds row m = lane%16; regs = K {half*8+0..7} then {16+half*8+0..7}
// B 32x16 bf16 (stored transposed N x K in LDS): lane holds col n = lane%16;
//   regs = K {half*8+0..7} then {16+half*8+0..7}, same as A.
// C/D 16x16 f32: VGPR r, lane l -> row r + 8*(l/16), col l%16.
// ---------------------------------------------------------------------------
__device__ __forceinline__ bf16x16 wmma_frag_a(const __bf16* base, int row, int stride,
                                               int kofs, int lane) {
  const __bf16* p = base + (size_t)(row + (lane & 15)) * stride + kofs + ((lane >> 4) << 3);
  union { uint4 q[2]; bf16x16 v; } u;
  u.q[0] = *(const uint4*)p;
  u.q[1] = *(const uint4*)(p + 16);
  return u.v;
}

__device__ __forceinline__ bf16x16 wmma_frag_b(const __bf16* base, int row, int stride,
                                               int kofs, int lane) {
  const __bf16* p = base + (size_t)(row + (lane & 15)) * stride + kofs + ((lane >> 4) << 3);
  union { uint4 q[2]; bf16x16 v; } u;
  u.q[0] = *(const uint4*)p;
  u.q[1] = *(const uint4*)(p + 16);
  return u.v;
}

__device__ __forceinline__ floatx8 wmma_bf16(bf16x16 a, bf16x16 b, floatx8 c) {
  return __builtin_amdgcn_wmma_f32_16x16x32_bf16(false, a, false, b, (short)0, c,
                                                 false, false);
}

__device__ __forceinline__ uint2 pack4bf(float a, float b, float c, float d) {
  union { __bf16 h[4]; uint2 u; } x;
  x.h[0] = (__bf16)a; x.h[1] = (__bf16)b; x.h[2] = (__bf16)c; x.h[3] = (__bf16)d;
  return x.u;
}

// gfx1250 async global->LDS copy (ASYNCcnt path, CDNA5 ISA)
__device__ __forceinline__ void async_load_b128(uint32_t lds_off, const void* gaddr) {
  asm volatile("global_load_async_to_lds_b128 %0, %1, off"
               :: "v"(lds_off), "v"(gaddr)
               : "memory");
}
__device__ __forceinline__ void wait_async() {
  asm volatile("s_wait_asynccnt 0" ::: "memory");
}

// 16-lane (half-wave) reductions with static XOR masks (stay within l/16 half)
__device__ __forceinline__ float hmax16(float v) {
  v = fmaxf(v, __shfl_xor(v, 1, 32));
  v = fmaxf(v, __shfl_xor(v, 2, 32));
  v = fmaxf(v, __shfl_xor(v, 4, 32));
  v = fmaxf(v, __shfl_xor(v, 8, 32));
  return v;
}
__device__ __forceinline__ float hsum16(float v) {
  v += __shfl_xor(v, 1, 32);
  v += __shfl_xor(v, 2, 32);
  v += __shfl_xor(v, 4, 32);
  v += __shfl_xor(v, 8, 32);
  return v;
}

// ---------------------------------------------------------------------------
// Tiled GEMM: C[M,N] = A[M,K] @ B[K,N] (+bias). A fp32 or bf16, B fp32.
// Block tile 128x128, K-step 32 (one WMMA K). 256 threads = 8 waves in 4x2 grid,
// each wave computes 2x4 = 8 16x16 tiles.
// ---------------------------------------------------------------------------
template <bool A_BF16, bool OUT_F32>
__global__ __launch_bounds__(256) void gemm_kernel(const void* __restrict__ Ap,
                                                   const float* __restrict__ B,
                                                   const float* __restrict__ bias,
                                                   void* __restrict__ Cp,
                                                   int M, int N, int K) {
  __shared__ __bf16 As[128][40];   // row-major, pad 40 -> conflict-free b128 frags
  __shared__ __bf16 Bs[128][40];   // B transposed: Bs[n][k]
  const int t = threadIdx.x;
  const int lane = t & 31, wave = t >> 5;
  const int m0 = blockIdx.y * 128, n0 = blockIdx.x * 128;
  const int wm = (wave >> 1) * 32;   // 0,32,64,96
  const int wn = (wave & 1) * 64;    // 0,64
  floatx8 acc[2][4] = {};

  for (int kt = 0; kt < K; kt += 32) {
    if constexpr (A_BF16) {
      const __bf16* A = (const __bf16*)Ap;
#pragma unroll
      for (int p = 0; p < 4; ++p) {
        int i = p * 32 + (t >> 3);
        int j = (t & 7) * 4;
        *(uint2*)&As[i][j] = *(const uint2*)(A + (size_t)(m0 + i) * K + kt + j);
      }
    } else {
      const float* A = (const float*)Ap;
#pragma unroll
      for (int p = 0; p < 4; ++p) {
        int i = p * 32 + (t >> 3);
        int j = (t & 7) * 4;
        float4 f = *(const float4*)(A + (size_t)(m0 + i) * K + kt + j);
        *(uint2*)&As[i][j] = pack4bf(f.x, f.y, f.z, f.w);
      }
    }
    // B tile 32 K-rows x 128 cols, transposed into Bs[n][k]
#pragma unroll
    for (int p = 0; p < 4; ++p) {
      int k = p * 8 + (t >> 5);
      int n = (t & 31) * 4;
      float4 f = *(const float4*)(B + (size_t)(kt + k) * N + n0 + n);
      Bs[n + 0][k] = (__bf16)f.x;
      Bs[n + 1][k] = (__bf16)f.y;
      Bs[n + 2][k] = (__bf16)f.z;
      Bs[n + 3][k] = (__bf16)f.w;
    }
    __syncthreads();

    bf16x16 a0 = wmma_frag_a(&As[0][0], wm,      40, 0, lane);
    bf16x16 a1 = wmma_frag_a(&As[0][0], wm + 16, 40, 0, lane);
#pragma unroll
    for (int nt = 0; nt < 4; ++nt) {
      bf16x16 b0 = wmma_frag_b(&Bs[0][0], wn + nt * 16, 40, 0, lane);
      acc[0][nt] = wmma_bf16(a0, b0, acc[0][nt]);
      acc[1][nt] = wmma_bf16(a1, b0, acc[1][nt]);
    }
    __syncthreads();
  }

  const int half8 = (lane >> 4) << 3;
#pragma unroll
  for (int mi = 0; mi < 2; ++mi)
#pragma unroll
    for (int nt = 0; nt < 4; ++nt) {
      int row = m0 + wm + mi * 16 + half8;
      int col = n0 + wn + nt * 16 + (lane & 15);
      if constexpr (OUT_F32) {
        float* C = (float*)Cp;
        float bv = bias[col];
#pragma unroll
        for (int r = 0; r < 8; ++r)
          C[(size_t)(row + r) * N + col] = acc[mi][nt][r] + bv;
      } else {
        __bf16* C = (__bf16*)Cp;
#pragma unroll
        for (int r = 0; r < 8; ++r)
          C[(size_t)(row + r) * N + col] = (__bf16)acc[mi][nt][r];
      }
    }
}

// ---------------------------------------------------------------------------
// Flash-style attention. One workgroup = 128 query rows of one (batch, head).
// qkv layout: [8192][3072] bf16, cols [Q | K | V], each h*64+d within its third.
// Per wave: 16-row strip, S = Q@K^T (16 WMMAs / KV block), online softmax
// (fully unrolled -> no M0/movrels indexing), P staged through LDS
// (C-layout -> A-layout), O += P@V (16 WMMAs / KV block).
// ---------------------------------------------------------------------------
__global__ __launch_bounds__(256) void attn_kernel(const void* __restrict__ qkvp,
                                                   void* __restrict__ outp) {
  const __bf16* qkv = (const __bf16*)qkvp;
  __bf16* out = (__bf16*)outp;
  __shared__ __bf16 Qs[128][72];      // 18 KB
  __shared__ __bf16 Ks[128][72];      // 18 KB  (B^T for Q@K^T: [j][d])
  __shared__ __bf16 Vt[64][136];      // 17 KB  (B^T for P@V: [d][j])
  __shared__ __bf16 Ps[8][16][136];   // 34 KB  per-wave P strips
  const int t = threadIdx.x, lane = t & 31, wave = t >> 5;
  const int b = (int)blockIdx.y >> 4, h = (int)blockIdx.y & 15;
  const int q0 = (int)blockIdx.x * 128;
  const size_t rs = QKVN;
  const __bf16* qbase = qkv + (size_t)b * SEQ * rs + h * HD;
  const __bf16* kbase = qbase + DIM;
  const __bf16* vbase = qbase + 2 * DIM;

#pragma unroll
  for (int p = 0; p < 8; ++p) {
    int i = p * 16 + (t >> 4);
    int c = (t & 15) * 4;
    *(uint2*)&Qs[i][c] = *(const uint2*)(qbase + (size_t)(q0 + i) * rs + c);
  }

  float mrow[8], lrow[8];
  floatx8 Oacc[4] = {};
#pragma unroll
  for (int r = 0; r < 8; ++r) { mrow[r] = -3.0e38f; lrow[r] = 0.f; }
  const int half8 = (lane >> 4) << 3;

  for (int jb = 0; jb < SEQ / 128; ++jb) {
    const int j0 = jb * 128;
    // K tile via async global->LDS (b128, 16B-aligned: pad stride 144B)
#pragma unroll
    for (int p = 0; p < 4; ++p) {
      int i = p * 32 + (t >> 3);
      int c = (t & 7) * 8;
      async_load_b128((uint32_t)(uintptr_t)&Ks[i][c],
                      kbase + (size_t)(j0 + i) * rs + c);
    }
    // V tile, stored transposed [d][j]
#pragma unroll
    for (int p = 0; p < 8; ++p) {
      int j = p * 16 + (t >> 4);
      int d0 = (t & 15) * 4;
      union { uint2 u; __bf16 hh[4]; } x;
      x.u = *(const uint2*)(vbase + (size_t)(j0 + j) * rs + d0);
      Vt[d0 + 0][j] = x.hh[0];
      Vt[d0 + 1][j] = x.hh[1];
      Vt[d0 + 2][j] = x.hh[2];
      Vt[d0 + 3][j] = x.hh[3];
    }
    wait_async();
    __syncthreads();

    // S = Q @ K^T for this wave's 16 rows x 128 kv cols
    floatx8 S[8] = {};
#pragma unroll
    for (int ds = 0; ds < HD; ds += 32) {
      bf16x16 a = wmma_frag_a(&Qs[0][0], wave * 16, 72, ds, lane);
#pragma unroll
      for (int jt = 0; jt < 8; ++jt) {
        bf16x16 bb = wmma_frag_b(&Ks[0][0], jt * 16, 72, ds, lane);
        S[jt] = wmma_bf16(a, bb, S[jt]);
      }
    }
#pragma unroll
    for (int jt = 0; jt < 8; ++jt) S[jt] *= 0.125f;   // HEAD_DIM^-0.5

    // online softmax; row (r + 8*half) lives in lanes of this half.
    // Fully unrolled: S[jt][r] with constant r -> static VGPR access (no movrels).
#pragma unroll
    for (int r = 0; r < 8; ++r) {
      float mx = S[0][r];
#pragma unroll
      for (int jt = 1; jt < 8; ++jt) mx = fmaxf(mx, S[jt][r]);
      mx = hmax16(mx);
      float mnew = fmaxf(mrow[r], mx);
      float fr = __expf(mrow[r] - mnew);
      float ssum = 0.f;
#pragma unroll
      for (int jt = 0; jt < 8; ++jt) {
        float pe = __expf(S[jt][r] - mnew);
        ssum += pe;
        Ps[wave][r + half8][jt * 16 + (lane & 15)] = (__bf16)pe;
      }
      ssum = hsum16(ssum);
      lrow[r] = lrow[r] * fr + ssum;
      mrow[r] = mnew;
#pragma unroll
      for (int dt = 0; dt < 4; ++dt) Oacc[dt][r] *= fr;
    }
    __syncthreads();   // Ps strip writes visible before P@V reads

    // O += P @ V  (P: 16x128 A-operand, Vt: B^T [d][j])
#pragma unroll
    for (int ks = 0; ks < 4; ++ks) {
      bf16x16 a = wmma_frag_a(&Ps[wave][0][0], 0, 136, ks * 32, lane);
#pragma unroll
      for (int dt = 0; dt < 4; ++dt) {
        bf16x16 bb = wmma_frag_b(&Vt[0][0], dt * 16, 136, ks * 32, lane);
        Oacc[dt] = wmma_bf16(a, bb, Oacc[dt]);
      }
    }
    __syncthreads();
  }

  // write O * (1/l) -> attn workspace [8192][1024] bf16 (row b*1024+n, col h*64+d)
  const size_t orow0 = (size_t)b * SEQ + q0 + wave * 16 + half8;
  float rl[8];
#pragma unroll
  for (int r = 0; r < 8; ++r) rl[r] = 1.0f / lrow[r];
#pragma unroll
  for (int dt = 0; dt < 4; ++dt) {
    int col = h * HD + dt * 16 + (lane & 15);
#pragma unroll
    for (int r = 0; r < 8; ++r)
      out[(orow0 + r) * DIM + col] = (__bf16)(Oacc[dt][r] * rl[r]);
  }
}

// ---------------------------------------------------------------------------
// Launch: QKV GEMM -> attention -> projection GEMM (+bias).
// Workspace: 48 MB bf16 QKV + 16 MB bf16 attention output = 64 MB.
// ---------------------------------------------------------------------------
extern "C" void kernel_launch(void* const* d_in, const int* in_sizes, int n_in,
                              void* d_out, int out_size, void* d_ws, size_t ws_size,
                              hipStream_t stream) {
  (void)in_sizes; (void)n_in; (void)out_size; (void)ws_size;
  const float* x      = (const float*)d_in[0];
  const float* w_qkv  = (const float*)d_in[1];
  const float* w_proj = (const float*)d_in[2];
  const float* b_proj = (const float*)d_in[3];
  float* out = (float*)d_out;

  void* qkv_ws  = d_ws;
  void* attn_ws = (void*)((char*)d_ws + (size_t)ROWS * QKVN * sizeof(__bf16));

  // 1) qkv = x @ w_qkv  (fp32 in, bf16 out)
  gemm_kernel<false, false><<<dim3(QKVN / 128, ROWS / 128), 256, 0, stream>>>(
      x, w_qkv, nullptr, qkv_ws, ROWS, QKVN, DIM);

  // 2) flash attention per (query block, batch*head)
  attn_kernel<<<dim3(SEQ / 128, BATCH * HEADS), 256, 0, stream>>>(qkv_ws, attn_ws);

  // 3) out = attn @ w_proj + b_proj  (bf16 in, fp32 out)
  gemm_kernel<true, true><<<dim3(DIM / 128, ROWS / 128), 256, 0, stream>>>(
      attn_ws, w_proj, b_proj, out, ROWS, DIM, DIM);
}
